// CompressedLookupTable_21328807592433
// MI455X (gfx1250) — compile-verified, matrix operand order checked by reading, not run on hardware
//
#include <hip/hip_runtime.h>

typedef __attribute__((ext_vector_type(8))) int v8i;

#define OUT_FEATURES 1024
#define NUM_K        256      // 64 blocks * 4 lattice dims
#define TABLE_DIM    4096

__device__ __forceinline__ int clampIdx(int x) {
    x = x < 0 ? 0 : x;
    return x > (TABLE_DIM - 1) ? (TABLE_DIM - 1) : x;
}

__global__ __launch_bounds__(256) void clt_wmma_kernel(
    const int*           __restrict__ ii,     // [512][256] row indices
    const int*           __restrict__ wi,     // [1024][256] col indices
    const unsigned char* __restrict__ table,  // [4096][4096]
    const float*         __restrict__ scale_p,
    const float*         __restrict__ zp_p,
    float*               __restrict__ out)    // [512][1024]
{
    __shared__ int lds_rowbase[NUM_K];        // clamp(ii[b,k]) * 4096, shared by block

    const int tid   = threadIdx.x;
    const int lane  = tid & 31;
    const int wave  = tid >> 5;
    const int wtask = blockIdx.x * 8 + wave;   // 0..32767
    const int b     = wtask >> 6;              // all 8 waves of a block share b
    const int o_base = (wtask & 63) << 4;      // 16-wide output tile
    const int m     = lane & 15;               // matrix row handled by this lane
    const int o     = o_base + m;              // lanes m and m+16 share the same o
    const int kofs  = (lane >> 4) << 3;        // upper half-wave covers K+8

    // Cooperative row-base build: one thread per k (256 threads, 256 k's).
    {
        int r = ii[b * NUM_K + tid];
        lds_rowbase[tid] = clampIdx(r) << 12;
    }
    __syncthreads();

    const int* __restrict__ wio = wi + o * NUM_K;

    v8i acc = {};
    v8i bones;
#pragma unroll
    for (int j = 0; j < 8; ++j) bones[j] = 0x01010101;  // B = ones(64x16) u8

#pragma unroll
    for (int kb = 0; kb < 4; ++kb) {          // 4 x 64-k accumulating WMMAs
        const int k0 = kb * 64;

        // ---- Phase 1: all 32 gather addresses for this 64-k block ----
        unsigned addr[32];
#pragma unroll
        for (int v = 0; v < 8; ++v) {
            // ISA 8-bit A-matrix 16x64 layout:
            // lanes 0-15 : v0:K0-3  v1:K4-7  v2:K16-19 v3:K20-23 v4-7:+32
            // lanes 16-31: same +8
            const int kstart = k0 + kofs
                             + ((v & 4) << 3)   // +32 for v>=4
                             + ((v & 2) << 3)   // +16 for odd pair
                             + ((v & 1) << 2);  // +4 within pair
            // ds_load_b128 broadcast: all lanes of a half-wave read same address
            const int4 rb = *(const int4*)(lds_rowbase + kstart);
            const int4 c4 = *(const int4*)(wio + kstart);  // 16B-aligned global
            addr[4 * v + 0] = (unsigned)(rb.x + clampIdx(c4.x));
            addr[4 * v + 1] = (unsigned)(rb.y + clampIdx(c4.y));
            addr[4 * v + 2] = (unsigned)(rb.z + clampIdx(c4.z));
            addr[4 * v + 3] = (unsigned)(rb.w + clampIdx(c4.w));
        }

        // Hard scheduling fence: nothing crosses. Forces all address math to
        // complete here, and prevents the gathers below from being pulled up
        // pairwise into the address phase.
        __builtin_amdgcn_sched_barrier(0);

        // ---- Phase 2: issue all 32 byte gathers as one batch ----
        // Half-wave lanes gather from the same 4KB table row per k -> L0/WGP$ hits.
        unsigned byteval[32];
#pragma unroll
        for (int t = 0; t < 32; ++t) byteval[t] = (unsigned)table[addr[t]];

        // Hard scheduling fence: the packs below may not be hoisted between the
        // gathers, so all 32 loads stay in flight and the waits become a
        // staged loadcnt ramp instead of per-pair full stalls.
        __builtin_amdgcn_sched_barrier(0);

        // ---- Phase 3: pack into the A-matrix dwords, then matrix-reduce ----
        v8i a;
#pragma unroll
        for (int v = 0; v < 8; ++v) {
            a[v] = (int)( byteval[4 * v + 0]
                       | (byteval[4 * v + 1] << 8)
                       | (byteval[4 * v + 2] << 16)
                       | (byteval[4 * v + 3] << 24));
        }
        // D[m,n] = sum_k A[m,k]*1 + C : 64-way k-reduction on the matrix unit
        acc = __builtin_amdgcn_wmma_i32_16x16x64_iu8(
                  /*sgn_a=*/false, a, /*sgn_b=*/false, bones, acc,
                  /*reuse_a=*/false, /*reuse_b=*/false);
    }

    const float scale = *scale_p;
    const float zp    = *zp_p;
    const float bias  = -(float)NUM_K * zp * scale;

    // C/D layout: VGPR j -> lanes 0-15 hold row M=j, lanes 16-31 hold row M=8+j,
    // replicated across all 16 columns. Lane 0 owns S0..S7, lane 16 owns S8..S15.
    if (m == 0) {
        float* op = out + b * OUT_FEATURES + o_base + ((lane >> 4) << 3);
        float4 f0, f1;
        f0.x = (float)acc[0] * scale + bias;
        f0.y = (float)acc[1] * scale + bias;
        f0.z = (float)acc[2] * scale + bias;
        f0.w = (float)acc[3] * scale + bias;
        f1.x = (float)acc[4] * scale + bias;
        f1.y = (float)acc[5] * scale + bias;
        f1.z = (float)acc[6] * scale + bias;
        f1.w = (float)acc[7] * scale + bias;
        *(float4*)(op)     = f0;
        *(float4*)(op + 4) = f1;
    }
}

extern "C" void kernel_launch(void* const* d_in, const int* in_sizes, int n_in,
                              void* d_out, int out_size, void* d_ws, size_t ws_size,
                              hipStream_t stream) {
    (void)in_sizes; (void)n_in; (void)d_ws; (void)ws_size; (void)out_size;
    const int*           ii    = (const int*)d_in[0];            // (512,64,4) int32
    const int*           wi    = (const int*)d_in[1];            // (1024,64,4) int32
    const unsigned char* table = (const unsigned char*)d_in[2];  // (4096,4096) uint8
    const float*         scale = (const float*)d_in[3];
    const float*         zp    = (const float*)d_in[4];
    float*               out   = (float*)d_out;                  // (512,1024) f32

    // 32768 wave-tasks (512 b x 64 o-tiles), 8 waves per 256-thread block
    dim3 grid(4096), block(256);
    clt_wmma_kernel<<<grid, block, 0, stream>>>(ii, wi, table, scale, zp, out);
}
